// ParticleTransformer_2551210574406
// MI455X (gfx1250) — matrix operand order repaired and checked by reading, not compiled, not run on hardware
//
#include <hip/hip_runtime.h>
#include <hip/hip_bf16.h>
#include <math.h>

#define EPSF 1e-8f

typedef __bf16 bf16;
typedef __attribute__((ext_vector_type(16))) __bf16 v16bf;
typedef __attribute__((ext_vector_type(8)))  float  v8f;

__device__ __forceinline__ v8f vzero8() {
  v8f z;
#pragma unroll
  for (int i = 0; i < 8; ++i) z[i] = 0.0f;
  return z;
}

__device__ __forceinline__ v8f wmma_bf16(v16bf a, v16bf b, v8f c) {
  // D = A(16x32 bf16) * B(32x16 bf16) + C(16x16 f32)
  return __builtin_amdgcn_wmma_f32_16x16x32_bf16(false, a, false, b, (short)0, c,
                                                 false, false);
}

// A fragment: A row-major [m][k], ld elems/row, base -> (m0, k0) of a 32-wide K chunk.
// Lane l<16 holds row m0+l, K {0..7,16..23}; lane l>=16 holds K {8..15,24..31}.
__device__ __forceinline__ v16bf frag_a(const bf16* base, int ld, int lane) {
  const bf16* p = base + (lane & 15) * ld + ((lane >> 4) << 3);
  v16bf f;
#pragma unroll
  for (int e = 0; e < 8; ++e) { f[e] = p[e]; f[e + 8] = p[e + 16]; }
  return f;
}

// B fragment: B^T stored [n][k], base -> (n0, k0). Lane l: col n0+(l&15),
// K {0..7,16..23} (l<16) or {8..15,24..31} (l>=16).
__device__ __forceinline__ v16bf frag_b(const bf16* base, int ld, int lane) {
  const bf16* p = base + (lane & 15) * ld + ((lane >> 4) << 3);
  v16bf f;
#pragma unroll
  for (int e = 0; e < 8; ++e) { f[e] = p[e]; f[e + 8] = p[e + 16]; }
  return f;
}

// Branch-free GELU (tanh form) built on the native transcendental unit
// (v_exp_f32 is a TRANS op that co-executes with WMMA); avoids the branchy
// device-lib erff expansion.
__device__ __forceinline__ float gelu_f(float x) {
  float u = 0.7978845608028654f * (x + 0.044715f * x * x * x);
  float e = __expf(-2.0f * fabsf(u));          // v_exp_f32, branch-free
  float t = (1.0f - e) / (1.0f + e);           // |tanh(u)|
  t = copysignf(t, u);
  return 0.5f * x * (1.0f + t);
}

// ---------------------------------------------------------------- weight prep
__global__ void k_w_transpose(const float* __restrict__ W, bf16* __restrict__ Wt,
                              int K, int N) {
  int idx = blockIdx.x * 256 + threadIdx.x;
  if (idx >= K * N) return;
  int n = idx / K, k = idx - n * K;
  Wt[idx] = (bf16)W[(size_t)k * N + n];   // Wt[n][k] = W[k][n]
}

// ---------------------------------------------------------------- small prologue
__global__ void k_ptjet(const float* __restrict__ x, float* __restrict__ ptjet) {
  int b = blockIdx.x, t = threadIdx.x;          // 128 threads
  float v = x[((size_t)b * 128 + t) * 7 + 1];   // pt
#pragma unroll
  for (int o = 16; o > 0; o >>= 1) v += __shfl_xor(v, o, 32);
  __shared__ float s[4];
  if ((t & 31) == 0) s[t >> 5] = v;
  __syncthreads();
  if (t == 0) ptjet[b] = s[0] + s[1] + s[2] + s[3];
}

__global__ void k_particle_pre(const float* __restrict__ x,
                               const float* __restrict__ ptjet,
                               float* __restrict__ pre) {
  int idx = blockIdx.x * 256 + threadIdx.x;     // < 4096
  const float* f = x + (size_t)idx * 7;
  int b = idx >> 7;
  float m_ = f[0], pt = f[1], eta = f[2], phi = f[3];
  float ch = coshf(eta), sh = sinhf(eta);
  float num = sqrtf(m_ * m_ + pt * pt * ch * ch) + pt * sh + EPSF;
  float den = sqrtf(m_ * m_ + pt * pt) + EPSF;
  float r = logf(num / den);
  float qs = f[6] * pt / (ptjet[b] + EPSF);
  float* o = pre + (size_t)idx * 8;
  o[0] = pt; o[1] = phi; o[2] = r; o[3] = qs; o[4] = f[4]; o[5] = f[5];
  o[6] = 0.f; o[7] = 0.f;
}

__global__ void k_ln_in(const float* __restrict__ x, const float* __restrict__ g,
                        const float* __restrict__ bb, float* __restrict__ out) {
  int idx = blockIdx.x * 256 + threadIdx.x;     // < 4096 rows of 7
  const float* r = x + (size_t)idx * 7;
  float s = 0.f;
#pragma unroll
  for (int f = 0; f < 7; ++f) s += r[f];
  float m = s * (1.0f / 7.0f);
  float vv = 0.f;
#pragma unroll
  for (int f = 0; f < 7; ++f) { float d = r[f] - m; vv += d * d; }
  float rs = rsqrtf(vv * (1.0f / 7.0f) + 1e-5f);
  float* o = out + (size_t)idx * 7;
#pragma unroll
  for (int f = 0; f < 7; ++f) o[f] = (r[f] - m) * rs * g[f] + bb[f];
}

__global__ void k_embed(const float* __restrict__ xln, const float* __restrict__ W,
                        const float* __restrict__ bb, float* __restrict__ out) {
  int idx = blockIdx.x * 256 + threadIdx.x;     // < 4096*128
  int row = idx >> 7, c = idx & 127;
  float a = bb[c];
#pragma unroll
  for (int f = 0; f < 7; ++f) a += xln[(size_t)row * 7 + f] * W[f * 128 + c];
  out[idx] = a;
}

// ---------------------------------------------------------------- LayerNorm C=128
__global__ void k_ln128(const float* __restrict__ in, float* __restrict__ out,
                        const float* __restrict__ g, const float* __restrict__ bb,
                        int M) {
  int wave = threadIdx.x >> 5, lane = threadIdx.x & 31;
  int row = blockIdx.x * 8 + wave;
  if (row >= M) return;
  const float* xr = in + (size_t)row * 128;
  float v0 = xr[lane], v1 = xr[lane + 32], v2 = xr[lane + 64], v3 = xr[lane + 96];
  float s = v0 + v1 + v2 + v3;
  float q = v0 * v0 + v1 * v1 + v2 * v2 + v3 * v3;
#pragma unroll
  for (int o = 16; o > 0; o >>= 1) { s += __shfl_xor(s, o, 32); q += __shfl_xor(q, o, 32); }
  float m = s * (1.0f / 128.0f);
  float var = q * (1.0f / 128.0f) - m * m;
  float rs = rsqrtf(var + 1e-5f);
  float* orow = out + (size_t)row * 128;
  orow[lane]      = (v0 - m) * rs * g[lane]      + bb[lane];
  orow[lane + 32] = (v1 - m) * rs * g[lane + 32] + bb[lane + 32];
  orow[lane + 64] = (v2 - m) * rs * g[lane + 64] + bb[lane + 64];
  orow[lane + 96] = (v3 - m) * rs * g[lane + 96] + bb[lane + 96];
}

// ---------------------------------------------------------------- generic WMMA GEMM
// out[M][N] = act(A[M][K] @ Bt[N][K]^T + bias) + resid. Block tile 128x64, 8 waves,
// each wave 32x32 (2x2 wmma). Full 128-wide K panel of A staged f32->bf16 in LDS
// (one barrier pair per panel, 16 wmma issued per panel per wave); B frags come
// straight from global (bf16 [N][K], L2-resident weights).
template <int ACT, bool BIAS, bool RESID>
__global__ void k_gemm(const float* __restrict__ A, const bf16* __restrict__ Bt,
                       const float* __restrict__ bias, const float* __restrict__ resid,
                       float* __restrict__ out, int M, int K, int N) {
  __shared__ bf16 As[128 * 136];   // 128 rows x 128 K panel, stride 136 (bank-safe)
  const int tid = threadIdx.x, lane = tid & 31, wave = tid >> 5;
  const int wm = wave & 3, wn = wave >> 2;
  const int bm = blockIdx.x * 128, bn = blockIdx.y * 64;

  v8f acc[2][2];
#pragma unroll
  for (int mi = 0; mi < 2; ++mi)
#pragma unroll
    for (int ni = 0; ni < 2; ++ni) acc[mi][ni] = vzero8();

  for (int k0 = 0; k0 < K; k0 += 128) {         // K is a multiple of 128 here
    __syncthreads();
#pragma unroll
    for (int it = 0; it < 16; ++it) {
      int idx = tid * 4 + it * 1024;
      int r = idx >> 7, c = idx & 127;
      int gr = bm + r;
      float4 v = make_float4(0.f, 0.f, 0.f, 0.f);
      if (gr < M) v = *(const float4*)(A + (size_t)gr * K + k0 + c);
      bf16* d = As + r * 136 + c;
      d[0] = (bf16)v.x; d[1] = (bf16)v.y; d[2] = (bf16)v.z; d[3] = (bf16)v.w;
    }
    __syncthreads();
#pragma unroll
    for (int kc = 0; kc < 4; ++kc) {
      v16bf a0 = frag_a(As + (wm * 32) * 136 + kc * 32, 136, lane);
      v16bf a1 = frag_a(As + (wm * 32 + 16) * 136 + kc * 32, 136, lane);
#pragma unroll
      for (int ni = 0; ni < 2; ++ni) {
        v16bf bb = frag_b(Bt + (size_t)(bn + wn * 32 + ni * 16) * K + k0 + kc * 32,
                          K, lane);
        acc[0][ni] = wmma_bf16(a0, bb, acc[0][ni]);
        acc[1][ni] = wmma_bf16(a1, bb, acc[1][ni]);
      }
    }
  }

  const int hi = (lane >> 4) << 3;
#pragma unroll
  for (int mi = 0; mi < 2; ++mi) {
#pragma unroll
    for (int ni = 0; ni < 2; ++ni) {
      int col = bn + wn * 32 + ni * 16 + (lane & 15);
      float bv = BIAS ? bias[col] : 0.0f;
#pragma unroll
      for (int v = 0; v < 8; ++v) {
        int row = bm + wm * 32 + mi * 16 + v + hi;
        if (row < M) {
          float xv = acc[mi][ni][v] + bv;
          if (ACT == 1) xv = gelu_f(xv);
          else if (ACT == 2) xv = fmaxf(xv, 0.f);
          if (RESID) xv += resid[(size_t)row * N + col];
          out[(size_t)row * N + col] = xv;
        }
      }
    }
  }
}

// ---------------------------------------------------------------- fused pairwise MLP
// One block per (b,i): 128 pairs. feats -> 7x128 VALU dense+gelu -> two 128x128
// WMMA dense layers in LDS -> 128x64 WMMA pair_proj (8 layers * 8 heads concat)
// straight to global biasAll. After the one barrier, each wave owns 16 rows.
__global__ void k_pairwise_fused(
    const float* __restrict__ pre, const unsigned char* __restrict__ mask,
    const float* __restrict__ W1, const float* __restrict__ b1,
    const bf16* __restrict__ W2t, const float* __restrict__ b2,
    const bf16* __restrict__ W3t, const float* __restrict__ b3,
    const bf16* __restrict__ WppT, const float* __restrict__ bpp,
    float* __restrict__ biasAll) {
  __shared__ __align__(16) char smem[34816 * 2];
  bf16* u1 = (bf16*)smem;                 // [128][136]
  bf16* u2 = (bf16*)(smem + 34816);       // [128][136]
  float* feat = (float*)(smem + 34816);   // overlays u2 head; dead before u2 written

  const int tid = threadIdx.x, lane = tid & 31, wave = tid >> 5;
  const int b = blockIdx.x >> 7, i = blockIdx.x & 127;

  // phase A: pair features of (i, j=tid) for j in [0,128)
  if (tid < 128) {
    const int j = tid;
    const float* pi = pre + (size_t)(b * 128 + i) * 8;
    const float* pj = pre + (size_t)(b * 128 + j) * 8;
    float pti = pi[0], phii = pi[1], ri = pi[2], qsi = pi[3], dxyi = pi[4], z0i = pi[5];
    float ptj = pj[0], phij = pj[1], rj = pj[2], qsj = pj[3], dxyj = pj[4], z0j = pj[5];
    float pm = (mask[b * 128 + i] && mask[b * 128 + j]) ? 1.0f : 0.0f;
    float dr = ri - rj, dphi = phii - phij;
    float delta = sqrtf(dr * dr + dphi * dphi + EPSF);
    float pmin = fminf(pti, ptj);
    float kt = pmin * delta;
    float zz = pmin / (pti + ptj + EPSF);
    float m2 = 2.0f * pti * ptj * (coshf(dr) - cosf(dphi)) + EPSF;
    float* fr = feat + j * 8;
    fr[0] = logf(fmaxf(delta, EPSF)) * pm;
    fr[1] = logf(fmaxf(kt, EPSF)) * pm;
    fr[2] = logf(fmaxf(zz, EPSF)) * pm;
    fr[3] = logf(fmaxf(m2, EPSF)) * pm;
    fr[4] = (dxyi - dxyj) * pm;
    fr[5] = (z0i - z0j) * pm;
    fr[6] = (qsi * qsj) * pm;
    fr[7] = 0.f;
  }
  __syncthreads();

  // phase B: u1 = gelu(feat @ W1 + b1)  (K=7, VALU)
#pragma unroll 4
  for (int it = 0; it < 64; ++it) {
    int idx = tid + it * 256;
    int j = idx >> 7, c = idx & 127;
    float a = b1[c];
#pragma unroll
    for (int f = 0; f < 7; ++f) a += feat[j * 8 + f] * W1[f * 128 + c];
    u1[j * 136 + c] = (bf16)gelu_f(a);
  }
  __syncthreads();   // last barrier: below, wave w owns rows [16w, 16w+16)

  const int m0 = wave * 16;
  const int hi = (lane >> 4) << 3;
  const int cl = lane & 15;

  // phase C: u2 = gelu(u1 @ W2 + b2)
#pragma unroll
  for (int nt = 0; nt < 8; ++nt) {
    v8f acc = vzero8();
#pragma unroll
    for (int kc = 0; kc < 4; ++kc) {
      v16bf a = frag_a(u1 + m0 * 136 + kc * 32, 136, lane);
      v16bf bb = frag_b(W2t + (size_t)(nt * 16) * 128 + kc * 32, 128, lane);
      acc = wmma_bf16(a, bb, acc);
    }
    int col = nt * 16 + cl;
    float bv = b2[col];
#pragma unroll
    for (int v = 0; v < 8; ++v)
      u2[(m0 + v + hi) * 136 + col] = (bf16)gelu_f(acc[v] + bv);
  }

  // phase D: u3 = u2 @ W3 + b3  (written back into u1 buffer, wave-local rows)
#pragma unroll
  for (int nt = 0; nt < 8; ++nt) {
    v8f acc = vzero8();
#pragma unroll
    for (int kc = 0; kc < 4; ++kc) {
      v16bf a = frag_a(u2 + m0 * 136 + kc * 32, 136, lane);
      v16bf bb = frag_b(W3t + (size_t)(nt * 16) * 128 + kc * 32, 128, lane);
      acc = wmma_bf16(a, bb, acc);
    }
    int col = nt * 16 + cl;
    float bv = b3[col];
#pragma unroll
    for (int v = 0; v < 8; ++v)
      u1[(m0 + v + hi) * 136 + col] = (bf16)(acc[v] + bv);
  }

  // phase E: biasAll[b,i,j, l*8+h] = u3 @ WppT + bpp  (N = 64 = 8 layers x 8 heads)
#pragma unroll
  for (int nt = 0; nt < 4; ++nt) {
    v8f acc = vzero8();
#pragma unroll
    for (int kc = 0; kc < 4; ++kc) {
      v16bf a = frag_a(u1 + m0 * 136 + kc * 32, 136, lane);
      v16bf bb = frag_b(WppT + (size_t)(nt * 16) * 128 + kc * 32, 128, lane);
      acc = wmma_bf16(a, bb, acc);
    }
    int col = nt * 16 + cl;
    float bv = bpp[col];
#pragma unroll
    for (int v = 0; v < 8; ++v) {
      int row = m0 + v + hi;
      biasAll[((size_t)(b * 128 + i) * 128 + row) * 64 + col] = acc[v] + bv;
    }
  }
}

// ---------------------------------------------------------------- qkv repack
// qkv f32 (B*N,384) layout [b][n][s][h][d] -> q/k/v bf16 [b][h][n][d]; scale on q.
__global__ void k_repack(const float* __restrict__ qkv, bf16* __restrict__ qh,
                         bf16* __restrict__ kh, bf16* __restrict__ vh) {
  int idx = blockIdx.x * 256 + threadIdx.x;     // < 4096*384
  int row = idx / 384, col = idx - row * 384;
  int sel = col >> 7, hh = (col >> 4) & 7, d = col & 15;
  int b = row >> 7, n = row & 127;
  float v = qkv[idx];
  size_t di = ((size_t)(b * 8 + hh) * 128 + n) * 16 + d;
  if (sel == 0)      qh[di] = (bf16)(v * 0.25f);   // fold 1/sqrt(D), D=16
  else if (sel == 1) kh[di] = (bf16)v;
  else               vh[di] = (bf16)v;
}

// ---------------------------------------------------------------- fused attention
// One block per (b,h): S = q k^T (WMMA, K=16 zero-padded to 32), +bias +mask,
// softmax in LDS (probs bf16 in place), O = P V (WMMA, K=128).
__global__ void k_attention(const bf16* __restrict__ qh, const bf16* __restrict__ kh,
                            const bf16* __restrict__ vh,
                            const float* __restrict__ biasAll,
                            const unsigned char* __restrict__ mask,
                            float* __restrict__ out, int layer) {
  __shared__ __align__(16) bf16 qs[128 * 40];
  __shared__ __align__(16) bf16 ks[128 * 40];
  __shared__ __align__(16) bf16 vt[16 * 136];
  __shared__ __align__(16) bf16 Ssm[128 * 136];

  const int tid = threadIdx.x, lane = tid & 31, wave = tid >> 5;
  const int bh = blockIdx.x, b = bh >> 3, h = bh & 7;
  const size_t base = (size_t)bh * 128 * 16;

  {
    uint4 qv = ((const uint4*)(qh + base))[tid];
    uint4 kv = ((const uint4*)(kh + base))[tid];
    int n = tid >> 1, dh = (tid & 1) << 3;
    *(uint4*)(qs + n * 40 + dh) = qv;
    *(uint4*)(ks + n * 40 + dh) = kv;
    uint4 z = make_uint4(0u, 0u, 0u, 0u);
    *(uint4*)(qs + n * 40 + 16 + dh) = z;      // zero K pad 16..31
    *(uint4*)(ks + n * 40 + 16 + dh) = z;
    const bf16* vsrc = vh + base;
#pragma unroll
    for (int e = 0; e < 8; ++e) {              // v[n][d] -> vt[d][n]
      int i2 = tid * 8 + e;
      vt[(i2 & 15) * 136 + (i2 >> 4)] = vsrc[i2];
    }
  }
  __syncthreads();

  const int m0 = wave * 16;
  const int hi = (lane >> 4) << 3;
  const int cl = lane & 15;

  {
    v16bf a = frag_a(qs + m0 * 40, 40, lane);
#pragma unroll
    for (int nt = 0; nt < 8; ++nt) {
      v16bf bb = frag_b(ks + nt * 16 * 40, 40, lane);
      v8f acc = wmma_bf16(a, bb, vzero8());
      int col = nt * 16 + cl;
#pragma unroll
      for (int v = 0; v < 8; ++v)
        Ssm[(m0 + v + hi) * 136 + col] = (bf16)acc[v];
    }
  }
  __syncthreads();

  if (tid < 128) {                              // softmax over row tid (query)
    const float* bptr =
        biasAll + (size_t)(b * 128 + tid) * 128 * 64 + (size_t)layer * 8 + h;
    bf16* row = Ssm + tid * 136;
    float mx = -INFINITY;
    for (int j = 0; j < 128; ++j) {
      float v = (float)row[j] + bptr[(size_t)j * 64];
      if (!mask[b * 128 + j]) v = -INFINITY;
      row[j] = (bf16)v;
      mx = fmaxf(mx, v);
    }
    float sum = 0.f;
    for (int j = 0; j < 128; ++j) {
      float e = __expf((float)row[j] - mx);     // native v_exp_f32
      sum += e;
      row[j] = (bf16)e;
    }
    float inv = 1.0f / sum;
    for (int j = 0; j < 128; ++j) row[j] = (bf16)((float)row[j] * inv);
  }
  __syncthreads();

  v8f acc = vzero8();                           // O = P @ V, one 16-col tile/wave
#pragma unroll
  for (int kc = 0; kc < 4; ++kc) {
    v16bf a = frag_a(Ssm + m0 * 136 + kc * 32, 136, lane);
    v16bf bb = frag_b(vt + kc * 32, 136, lane);
    acc = wmma_bf16(a, bb, acc);
  }
#pragma unroll
  for (int v = 0; v < 8; ++v)
    out[((size_t)b * 128 + (m0 + v + hi)) * 128 + h * 16 + cl] = acc[v];
}

// ---------------------------------------------------------------- cls path
__global__ void k_concat(const float* __restrict__ x, const float* __restrict__ cls,
                         float* __restrict__ xc) {
  int idx = blockIdx.x * 256 + threadIdx.x;     // < 32*129*128
  int b = idx / (129 * 128);
  int r = idx - b * 129 * 128;
  int n = r >> 7, c = r & 127;
  xc[idx] = (n == 0) ? cls[c] : x[((size_t)b * 128 + n - 1) * 128 + c];
}

__global__ void k_cls_attn(const float* __restrict__ hln, const float* __restrict__ kvbuf,
                           const float* __restrict__ Wq, const float* __restrict__ Wproj,
                           const float* __restrict__ bproj,
                           const unsigned char* __restrict__ mask,
                           float* __restrict__ xc) {
  int b = blockIdx.x, t = threadIdx.x;          // 256 threads
  __shared__ float h0[128], qc[128], sc[8 * 132], ov[128];
  if (t < 128) h0[t] = hln[(size_t)(b * 129) * 128 + t];
  __syncthreads();
  if (t < 128) {
    float a = 0.f;
    for (int k = 0; k < 128; ++k) a += h0[k] * Wq[k * 128 + t];
    qc[t] = a;
  }
  __syncthreads();
  for (int idx = t; idx < 8 * 129; idx += 256) {
    int hh = idx / 129, j = idx - hh * 129;
    bool ok = (j == 0) || (mask[b * 128 + j - 1] != 0);
    float s = -INFINITY;
    if (ok) {
      s = 0.f;
      const float* kr = kvbuf + ((size_t)(b * 129 + j)) * 256 + hh * 16;
      for (int d = 0; d < 16; ++d) s += qc[hh * 16 + d] * kr[d];
      s *= 0.25f;
    }
    sc[hh * 132 + j] = s;
  }
  __syncthreads();
  if (t < 8) {
    float mx = -INFINITY;
    for (int j = 0; j < 129; ++j) mx = fmaxf(mx, sc[t * 132 + j]);
    float sum = 0.f;
    for (int j = 0; j < 129; ++j) { float e = __expf(sc[t * 132 + j] - mx); sc[t * 132 + j] = e; sum += e; }
    float inv = 1.0f / sum;
    for (int j = 0; j < 129; ++j) sc[t * 132 + j] *= inv;
  }
  __syncthreads();
  if (t < 128) {
    int hh = t >> 4, d = t & 15;
    float a = 0.f;
    for (int j = 0; j < 129; ++j)
      a += sc[hh * 132 + j] * kvbuf[((size_t)(b * 129 + j)) * 256 + 128 + hh * 16 + d];
    ov[t] = a;
  }
  __syncthreads();
  if (t < 128) {
    float a = bproj[t];
    for (int k = 0; k < 128; ++k) a += ov[k] * Wproj[k * 128 + t];
    xc[(size_t)(b * 129) * 128 + t] += a;       // new_cls = res[:,0] + proj(o)
  }
}

__global__ void k_head(const float* __restrict__ xc, const float* __restrict__ gw,
                       const float* __restrict__ gb, const float* __restrict__ w0,
                       const float* __restrict__ b0, const float* __restrict__ w1,
                       const float* __restrict__ b1, float* __restrict__ out) {
  int b = blockIdx.x, t = threadIdx.x;          // 128 threads
  __shared__ float cbuf[128], s1[4], s2[4];
  float v = xc[(size_t)(b * 129) * 128 + t];
  float a = v, q = v * v;
#pragma unroll
  for (int o = 16; o > 0; o >>= 1) { a += __shfl_xor(a, o, 32); q += __shfl_xor(q, o, 32); }
  if ((t & 31) == 0) { s1[t >> 5] = a; s2[t >> 5] = q; }
  __syncthreads();
  float S1 = s1[0] + s1[1] + s1[2] + s1[3];
  float S2 = s2[0] + s2[1] + s2[2] + s2[3];
  float m = S1 * (1.0f / 128.0f);
  float var = S2 * (1.0f / 128.0f) - m * m;
  cbuf[t] = (v - m) * rsqrtf(var + 1e-5f) * gw[t] + gb[t];
  __syncthreads();
  float h = b0[t];
  for (int k = 0; k < 128; ++k) h += cbuf[k] * w0[k * 128 + t];
  h = fmaxf(h, 0.f);
  float p = h * w1[t];
#pragma unroll
  for (int o = 16; o > 0; o >>= 1) p += __shfl_xor(p, o, 32);
  __syncthreads();
  if ((t & 31) == 0) s1[t >> 5] = p;
  __syncthreads();
  if (t == 0) out[b] = s1[0] + s1[1] + s1[2] + s1[3] + b1[0];
}

// ================================================================ host
extern "C" void kernel_launch(void* const* d_in, const int* in_sizes, int n_in,
                              void* d_out, int out_size, void* d_ws, size_t ws_size,
                              hipStream_t stream) {
  (void)in_sizes; (void)n_in; (void)out_size; (void)ws_size;
  const float* x = (const float*)d_in[0];
  const unsigned char* mask = (const unsigned char*)d_in[1];
  auto F32 = [&](int i) { return (const float*)d_in[i]; };
  // insertion-order flattening: 0 x, 1 mask, 2-3 ln_in, 4-5 embed, 6-11 pair[0..2],
  // 12 cls_token, 13.. pblocks (13 leaves each), 117.. cblocks (12 each),
  // 141-142 norm, 143-146 head.
  const int PB0 = 13, CB0 = PB0 + 8 * 13;
  auto pb = [&](int l, int j) { return (const float*)d_in[PB0 + l * 13 + j]; };
  auto cb = [&](int i, int j) { return (const float*)d_in[CB0 + i * 12 + j]; };

  size_t off = 0;
  auto alloc = [&](size_t bytes) -> void* {
    void* p = (char*)d_ws + off;
    off += (bytes + 255) & ~(size_t)255;
    return p;
  };

  float* ptjet = (float*)alloc(32 * 4);
  float* pre   = (float*)alloc((size_t)4096 * 8 * 4);
  float* xln   = (float*)alloc((size_t)4096 * 7 * 4);
  float* xA    = (float*)alloc((size_t)4096 * 128 * 4);
  float* xB    = (float*)alloc((size_t)4096 * 128 * 4);
  float* hbuf  = (float*)alloc((size_t)4096 * 128 * 4);
  float* qkvb  = (float*)alloc((size_t)4096 * 384 * 4);
  bf16*  qhb   = (bf16*)alloc((size_t)524288 * 2);
  bf16*  khb   = (bf16*)alloc((size_t)524288 * 2);
  bf16*  vhb   = (bf16*)alloc((size_t)524288 * 2);
  float* obuf  = (float*)alloc((size_t)4096 * 128 * 4);
  float* mbuf  = (float*)alloc((size_t)4096 * 512 * 4);
  float* xcA   = (float*)alloc((size_t)4128 * 128 * 4);
  float* xcB   = (float*)alloc((size_t)4128 * 128 * 4);
  float* hcb   = (float*)alloc((size_t)4128 * 128 * 4);
  float* kvb   = (float*)alloc((size_t)4128 * 256 * 4);
  float* mcb   = (float*)alloc((size_t)4128 * 512 * 4);
  float* biasAll = (float*)alloc((size_t)524288 * 64 * 4);   // 134 MB

  bf16* W2t  = (bf16*)alloc((size_t)128 * 128 * 2);
  bf16* W3t  = (bf16*)alloc((size_t)128 * 128 * 2);
  bf16* WppT = (bf16*)alloc((size_t)64 * 128 * 2);
  float* bpp = (float*)alloc(64 * 4);
  bf16 *WqkvT[8], *WprojT[8], *Wm1T[8], *Wm2T[8];
  for (int l = 0; l < 8; ++l) {
    WqkvT[l]  = (bf16*)alloc((size_t)384 * 128 * 2);
    WprojT[l] = (bf16*)alloc((size_t)128 * 128 * 2);
    Wm1T[l]   = (bf16*)alloc((size_t)512 * 128 * 2);
    Wm2T[l]   = (bf16*)alloc((size_t)128 * 512 * 2);
  }
  bf16 *WkvT[2], *Wm1cT[2], *Wm2cT[2];
  for (int i = 0; i < 2; ++i) {
    WkvT[i]  = (bf16*)alloc((size_t)256 * 128 * 2);
    Wm1cT[i] = (bf16*)alloc((size_t)512 * 128 * 2);
    Wm2cT[i] = (bf16*)alloc((size_t)128 * 512 * 2);
  }

  auto transpose = [&](const float* W, bf16* Wt, int K, int N) {
    int total = K * N;
    k_w_transpose<<<dim3((total + 255) / 256), dim3(256), 0, stream>>>(W, Wt, K, N);
  };

  // weight prep: f32 [K][N] -> bf16 [N][K]
  transpose(F32(8),  W2t, 128, 128);   // pair[1].w
  transpose(F32(10), W3t, 128, 128);   // pair[2].w
  for (int l = 0; l < 8; ++l) {
    transpose(pb(l, 3), WppT + (size_t)l * 8 * 128, 128, 8);  // pair_proj cat rows l*8..
    hipMemcpyAsync(bpp + l * 8, pb(l, 4), 8 * sizeof(float),
                   hipMemcpyDeviceToDevice, stream);
    transpose(pb(l, 0), WqkvT[l], 128, 384);
    transpose(pb(l, 1), WprojT[l], 128, 128);
    transpose(pb(l, 5), Wm1T[l], 128, 512);
    transpose(pb(l, 7), Wm2T[l], 512, 128);
  }
  for (int i = 0; i < 2; ++i) {
    transpose(cb(i, 1), WkvT[i], 128, 256);
    transpose(cb(i, 4), Wm1cT[i], 128, 512);
    transpose(cb(i, 6), Wm2cT[i], 512, 128);
  }

  // prologue
  k_ptjet<<<dim3(32), dim3(128), 0, stream>>>(x, ptjet);
  k_particle_pre<<<dim3(16), dim3(256), 0, stream>>>(x, ptjet, pre);
  k_ln_in<<<dim3(16), dim3(256), 0, stream>>>(x, F32(2), F32(3), xln);
  k_embed<<<dim3(2048), dim3(256), 0, stream>>>(xln, F32(4), F32(5), xA);

  // pairwise MLP fused in LDS -> per-layer/head attention bias
  k_pairwise_fused<<<dim3(4096), dim3(256), 0, stream>>>(
      pre, mask, F32(6), F32(7), W2t, F32(9), W3t, F32(11), WppT, bpp, biasAll);

  auto gemm = [&](void (*kern)(const float*, const bf16*, const float*, const float*,
                               float*, int, int, int),
                  const float* A, const bf16* Bt, const float* bias, const float* resid,
                  float* outp, int M, int K, int N) {
    dim3 grid((M + 127) / 128, N / 64);
    kern<<<grid, dim3(256), 0, stream>>>(A, Bt, bias, resid, outp, M, K, N);
  };

  // particle blocks
  for (int l = 0; l < 8; ++l) {
    k_ln128<<<dim3(512), dim3(256), 0, stream>>>(xA, hbuf, pb(l, 9), pb(l, 10), 4096);
    gemm(k_gemm<0, false, false>, hbuf, WqkvT[l], nullptr, nullptr, qkvb, 4096, 128, 384);
    k_repack<<<dim3(6144), dim3(256), 0, stream>>>(qkvb, qhb, khb, vhb);
    k_attention<<<dim3(256), dim3(256), 0, stream>>>(qhb, khb, vhb, biasAll, mask, obuf, l);
    gemm(k_gemm<0, true, true>, obuf, WprojT[l], pb(l, 2), xA, xB, 4096, 128, 128);
    k_ln128<<<dim3(512), dim3(256), 0, stream>>>(xB, hbuf, pb(l, 11), pb(l, 12), 4096);
    gemm(k_gemm<1, true, false>, hbuf, Wm1T[l], pb(l, 6), nullptr, mbuf, 4096, 128, 512);
    gemm(k_gemm<0, true, true>, mbuf, Wm2T[l], pb(l, 8), xB, xA, 4096, 512, 128);
  }

  // class-attention blocks
  k_concat<<<dim3(2064), dim3(256), 0, stream>>>(xA, F32(12), xcA);
  float* cin = xcA; float* cout = xcB;
  for (int i = 0; i < 2; ++i) {
    k_ln128<<<dim3(516), dim3(256), 0, stream>>>(cin, hcb, cb(i, 8), cb(i, 9), 4128);
    gemm(k_gemm<0, false, false>, hcb, WkvT[i], nullptr, nullptr, kvb, 4128, 128, 256);
    k_cls_attn<<<dim3(32), dim3(256), 0, stream>>>(hcb, kvb, cb(i, 0), cb(i, 2), cb(i, 3),
                                                   mask, cin);
    k_ln128<<<dim3(516), dim3(256), 0, stream>>>(cin, hcb, cb(i, 10), cb(i, 11), 4128);
    gemm(k_gemm<1, true, false>, hcb, Wm1cT[i], cb(i, 5), nullptr, mcb, 4128, 128, 512);
    gemm(k_gemm<0, true, true>, mcb, Wm2cT[i], cb(i, 7), cin, cout, 4128, 512, 128);
    float* t = cin; cin = cout; cout = t;
  }

  // final norm (row 0 only matters) + MLP head
  k_head<<<dim3(32), dim3(128), 0, stream>>>(cin, F32(141), F32(142), F32(143), F32(144),
                                             F32(145), F32(146), (float*)d_out);
}
